// UKT_40836549050929
// MI455X (gfx1250) — matrix operand order repaired and checked by reading, not compiled, not run on hardware
//
#include <hip/hip_runtime.h>
#include <hip/hip_bf16.h>

// ---------------------------------------------------------------------------
// CDNA5 (gfx1250) wave32 WMMA + async-to-LDS helpers
// ---------------------------------------------------------------------------
typedef __attribute__((ext_vector_type(16))) __bf16 v16bf;
typedef __attribute__((ext_vector_type(8)))  float  v8f;

#define WMMA_BF16(a, b, c) \
  __builtin_amdgcn_wmma_f32_16x16x32_bf16(false, (a), false, (b), (short)0, (c), false, false)

// Low 32 bits of a flat shared-memory pointer == LDS byte offset.
__device__ __forceinline__ unsigned lds_off(const void* p) {
  return (unsigned)(unsigned long long)(uintptr_t)p;
}

// GLOBAL_LOAD_ASYNC_TO_LDS_B128: per-lane 16B global->LDS copy, ASYNCcnt-tracked.
__device__ __forceinline__ void async_cp16(unsigned ldsAddr, const void* g) {
  asm volatile("global_load_async_to_lds_b128 %0, %1, off"
               :
               : "v"(ldsAddr), "v"((unsigned long long)(uintptr_t)g)
               : "memory");
}
__device__ __forceinline__ void async_wait0() {
  asm volatile("s_wait_asynccnt 0x0" ::: "memory");
}

// A-matrix 16x32 bf16 fragment from LDS tile [rows][ldk].
// lane m = lane&15; hi half-wave offsets K by +8 per half.
__device__ __forceinline__ v16bf ld_a_bf16(const __bf16* base, int mrow, int ldk,
                                           int k0, int lane) {
  const int m  = mrow + (lane & 15);
  const int hi = (lane >> 4) * 8;
  const __bf16* p = base + m * ldk + k0 + hi;
  v16bf a;
#pragma unroll
  for (int i = 0; i < 8; ++i) {
    const int kk = (i < 4) ? (2 * i) : (8 + 2 * i);  // 0,2,4,6,16,18,20,22
    a[2 * i]     = p[kk];
    a[2 * i + 1] = p[kk + 1];
  }
  return a;
}

// B-matrix 32x16 bf16 fragment, K contiguous in LDS: tile layout [n][ldn] rows.
// n = n0 + (lane&15); K = k0 + (lane>=16?8:0) + (j<8 ? j : j+8).
__device__ __forceinline__ v16bf ld_b_bf16(const __bf16* base, int n0, int ldn,
                                           int k0, int lane) {
  const int n  = n0 + (lane & 15);
  const int kh = k0 + (lane >> 4) * 8;
  const __bf16* p = base + n * ldn + kh;
  v16bf b;
#pragma unroll
  for (int j = 0; j < 16; ++j) b[j] = p[(j < 8) ? j : (j + 8)];
  return b;
}

// A fragment from an fp32 LDS panel (attention probabilities), optional square.
__device__ __forceinline__ v16bf ld_a_f32(const float* base, int mrow, int ld,
                                          int k0, int lane, bool sq) {
  const int m  = mrow + (lane & 15);
  const int hi = (lane >> 4) * 8;
  const float* p = base + m * ld + k0 + hi;
  v16bf a;
#pragma unroll
  for (int i = 0; i < 8; ++i) {
    const int kk = (i < 4) ? (2 * i) : (8 + 2 * i);
    float x0 = p[kk], x1 = p[kk + 1];
    if (sq) { x0 *= x0; x1 *= x1; }
    a[2 * i]     = (__bf16)x0;
    a[2 * i + 1] = (__bf16)x1;
  }
  return a;
}

// ---------------------------------------------------------------------------
// C = A @ W^T ; A:[8192,512] (f32 or bf16), W:[512,512] f32.
// OUT_BHSD: bf16 into [b,h,s,d]; else f32 row-major.
// 256 threads = 8 waves -> 64x64 tile, wave -> 16x32 sub-tile.
// LDS tiles padded to 40 bf16/row (80B): conflict-free fragment reads.
// ---------------------------------------------------------------------------
template <bool A_BF16, bool OUT_BHSD>
__global__ __launch_bounds__(256) void gemm_awt_kernel(const void* __restrict__ Av,
                                                       const float* __restrict__ W,
                                                       void* __restrict__ Outv) {
  __shared__ __bf16 lA[64 * 40];
  __shared__ __bf16 lB[64 * 40];
  const int tid = threadIdx.x, lane = tid & 31, w = tid >> 5;
  const int bm = blockIdx.x, bn = blockIdx.y;
  const int mrow = (w & 3) * 16, ncol = (w >> 2) * 32;
  v8f acc0 = {0.f, 0.f, 0.f, 0.f, 0.f, 0.f, 0.f, 0.f};
  v8f acc1 = acc0;
  const int srow = tid >> 2;        // 0..63
  const int scg  = (tid & 3) * 8;   // element col group
  const unsigned laOff = lds_off(lA);

  for (int k0 = 0; k0 < 512; k0 += 32) {
    if constexpr (A_BF16) {
      // async raw copy: thread -> one 16B chunk (8 bf16)
      const char* g =
          (const char*)((const __bf16*)Av + (size_t)(bm * 64 + srow) * 512 + k0 + scg);
      async_cp16(laOff + (unsigned)(srow * 80 + scg * 2), g);
    } else {
      const float* ap = (const float*)Av + (size_t)(bm * 64 + srow) * 512 + k0 + scg;
      __builtin_prefetch(ap + 32, 0, 0);
#pragma unroll
      for (int t = 0; t < 8; ++t) lA[srow * 40 + scg + t] = (__bf16)ap[t];
    }
    const float* wp = W + (size_t)(bn * 64 + srow) * 512 + k0 + scg;
    __builtin_prefetch(wp + 32, 0, 0);
#pragma unroll
    for (int t = 0; t < 8; ++t) lB[srow * 40 + scg + t] = (__bf16)wp[t];
    if constexpr (A_BF16) async_wait0();
    __syncthreads();

    v16bf a  = ld_a_bf16(lA, mrow, 40, 0, lane);
    v16bf b0 = ld_b_bf16(lB, ncol,      40, 0, lane);
    v16bf b1 = ld_b_bf16(lB, ncol + 16, 40, 0, lane);
    acc0 = WMMA_BF16(a, b0, acc0);
    acc1 = WMMA_BF16(a, b1, acc1);
    __syncthreads();
  }

  const int hi8 = (lane >> 4) * 8, ln = lane & 15;
#pragma unroll
  for (int nt = 0; nt < 2; ++nt) {
    const v8f& acc = nt ? acc1 : acc0;
#pragma unroll
    for (int r = 0; r < 8; ++r) {
      const int gm = bm * 64 + mrow + r + hi8;
      const int gn = bn * 64 + ncol + nt * 16 + ln;
      if constexpr (OUT_BHSD) {
        const int b = gm >> 9, s = gm & 511, h = gn >> 6, d = gn & 63;
        ((__bf16*)Outv)[(((size_t)b * 8 + h) * 512 + s) * 64 + d] = (__bf16)acc[r];
      } else {
        ((float*)Outv)[(size_t)gm * 512 + gn] = acc[r];
      }
    }
  }
}

// ---------------------------------------------------------------------------
// Row stats over last dim (64): mode 0 -> rs = sum(v^2); mode 1 -> rs = sum(v)
// (raw cov sum) and in-place rewrite with sqrt(max(v,1e-24)). Wave per row.
// ---------------------------------------------------------------------------
__global__ __launch_bounds__(256) void stats_kernel(__bf16* __restrict__ X,
                                                    float* __restrict__ rs, int mode) {
  const int wv = threadIdx.x >> 5, lane = threadIdx.x & 31;
  const size_t row = (size_t)blockIdx.x * 8 + wv;
  __bf16* p = X + row * 64;
  const float v0 = (float)p[lane], v1 = (float)p[lane + 32];
  float part = (mode == 0) ? (v0 * v0 + v1 * v1) : (v0 + v1);
  for (int off = 16; off; off >>= 1) part += __shfl_xor(part, off, 32);
  if (mode == 1) {
    p[lane]      = (__bf16)sqrtf(fmaxf(v0, 1e-24f));
    p[lane + 32] = (__bf16)sqrtf(fmaxf(v1, 1e-24f));
  }
  if (lane == 0) rs[row] = part;
}

// ---------------------------------------------------------------------------
// Fused Wasserstein attention. One block = (b, h, 64-query-row panel).
// LDS: 64x516 f32 panel + 2 Q tiles + 2 K/V tiles (64x72 bf16) = 168960 B.
// Q/K tiles staged with GLOBAL_LOAD_ASYNC_TO_LDS_B128; V staged transposed.
// ---------------------------------------------------------------------------
#define SLD 516
#define TLD 72
#define VLD 40

__global__ __launch_bounds__(256) void attn_kernel(
    const __bf16* __restrict__ QM, const __bf16* __restrict__ SQC,
    const __bf16* __restrict__ KM, const __bf16* __restrict__ SKC,
    const __bf16* __restrict__ VM, const __bf16* __restrict__ VC,
    const float* __restrict__ RSqm, const float* __restrict__ RSqc,
    const float* __restrict__ RSkm, const float* __restrict__ RSkc,
    const float* __restrict__ gammas, __bf16* __restrict__ CM,
    __bf16* __restrict__ CC) {
  extern __shared__ char smem[];
  float*  S    = (float*)smem;                           // [64][516]
  __bf16* lQM  = (__bf16*)(smem + 64 * SLD * 4);         // [64][72]
  __bf16* lSQC = lQM + 64 * TLD;
  __bf16* lK1  = lSQC + 64 * TLD;
  __bf16* lK2  = lK1 + 64 * TLD;

  const int tid = threadIdx.x, lane = tid & 31, w = tid >> 5;
  const int rb = blockIdx.x & 7, h = (blockIdx.x >> 3) & 7, b = blockIdx.x >> 6;
  const size_t headOff = ((size_t)b * 8 + h) * 512 * 64;
  const size_t rsOff   = ((size_t)b * 8 + h) * 512;
  const int mrow = (w & 3) * 16, ncol = (w >> 2) * 32;
  const int hi8 = (lane >> 4) * 8, ln = lane & 15;

  const unsigned oQM = lds_off(lQM), oSQC = lds_off(lSQC);
  const unsigned oK1 = lds_off(lK1), oK2 = lds_off(lK2);

  // ---- stage Q tiles via async copies (each: 512 chunks of 16B) ----
  {
    const char* g1 = (const char*)(QM + headOff + (size_t)rb * 64 * 64);
    const char* g2 = (const char*)(SQC + headOff + (size_t)rb * 64 * 64);
    for (int i = tid; i < 512; i += 256) {
      const unsigned l = (unsigned)((i >> 2) * (TLD * 2) + (i & 3) * 16);
      async_cp16(oQM + l, g1 + i * 16);
      async_cp16(oSQC + l, g2 + i * 16);
    }
    async_wait0();
  }
  __syncthreads();

  // hoisted Q fragments + per-row q stats (invariant over key tiles)
  v16bf aM0 = ld_a_bf16(lQM, mrow, TLD, 0, lane);
  v16bf aM1 = ld_a_bf16(lQM, mrow, TLD, 32, lane);
  v16bf aC0 = ld_a_bf16(lSQC, mrow, TLD, 0, lane);
  v16bf aC1 = ld_a_bf16(lSQC, mrow, TLD, 32, lane);
  float qms[8], qcs[8];
#pragma unroll
  for (int r = 0; r < 8; ++r) {
    const int m = mrow + r + hi8;
    qms[r] = RSqm[rsOff + rb * 64 + m];
    qcs[r] = RSqc[rsOff + rb * 64 + m];
  }

  // ---- phase 1: score panel via WMMA (mean + cov paths) ----
  const char* gkm = (const char*)(KM + headOff);
  const char* gkc = (const char*)(SKC + headOff);
  for (int kt = 0; kt < 8; ++kt) {
    __syncthreads();  // protect lK1/lK2 reuse
    for (int i = tid; i < 512; i += 256) {
      const unsigned l = (unsigned)((i >> 2) * (TLD * 2) + (i & 3) * 16);
      async_cp16(oK1 + l, gkm + (size_t)kt * 8192 + i * 16);
      async_cp16(oK2 + l, gkc + (size_t)kt * 8192 + i * 16);
    }
    async_wait0();
    __syncthreads();

    v8f am0 = {0.f, 0.f, 0.f, 0.f, 0.f, 0.f, 0.f, 0.f};
    v8f am1 = am0, ac0 = am0, ac1 = am0;
    {
      v16bf bM0 = ld_b_bf16(lK1, ncol,      TLD, 0, lane);
      v16bf bM1 = ld_b_bf16(lK1, ncol + 16, TLD, 0, lane);
      v16bf bC0 = ld_b_bf16(lK2, ncol,      TLD, 0, lane);
      v16bf bC1 = ld_b_bf16(lK2, ncol + 16, TLD, 0, lane);
      am0 = WMMA_BF16(aM0, bM0, am0);
      am1 = WMMA_BF16(aM0, bM1, am1);
      ac0 = WMMA_BF16(aC0, bC0, ac0);
      ac1 = WMMA_BF16(aC0, bC1, ac1);
      bM0 = ld_b_bf16(lK1, ncol,      TLD, 32, lane);
      bM1 = ld_b_bf16(lK1, ncol + 16, TLD, 32, lane);
      bC0 = ld_b_bf16(lK2, ncol,      TLD, 32, lane);
      bC1 = ld_b_bf16(lK2, ncol + 16, TLD, 32, lane);
      am0 = WMMA_BF16(aM1, bM0, am0);
      am1 = WMMA_BF16(aM1, bM1, am1);
      ac0 = WMMA_BF16(aC1, bC0, ac0);
      ac1 = WMMA_BF16(aC1, bC1, ac1);
    }
    // score = -((|q|^2+|k|^2-2qk) + (Sq+Sk-2*sq*sk)) / sqrt(64)
#pragma unroll
    for (int nt = 0; nt < 2; ++nt) {
      const int kg = kt * 64 + ncol + nt * 16 + ln;
      const float kms = RSkm[rsOff + kg], kcs = RSkc[rsOff + kg];
      const v8f& am = nt ? am1 : am0;
      const v8f& ac = nt ? ac1 : ac0;
#pragma unroll
      for (int r = 0; r < 8; ++r) {
        const int m = mrow + r + hi8;
        const float sc =
            -((qms[r] + kms - 2.f * am[r]) + (qcs[r] + kcs - 2.f * ac[r])) * 0.125f;
        S[m * SLD + kg] = sc;
      }
    }
  }
  __syncthreads();

  // ---- phase 2: masked softmax -> cumsum -> distance decay -> softmax ----
  {
    const float g  = gammas[h];
    const float sp = (g > 20.f) ? g : log1pf(__expf(g));  // softplus
    const float gamma = -sp;
    for (int rr = w; rr < 64; rr += 8) {
      const int ig = rb * 64 + rr;  // global query index
      float* row = S + rr * SLD;
      const int j0 = lane * 16;
      float x[16];
#pragma unroll
      for (int t = 0; t < 16; ++t) x[t] = row[j0 + t];

      float mx = -3.0e38f;
#pragma unroll
      for (int t = 0; t < 16; ++t)
        if (j0 + t <= ig) mx = fmaxf(mx, x[t]);
      for (int off = 16; off; off >>= 1) mx = fmaxf(mx, __shfl_xor(mx, off, 32));

      float e[16], c = 0.f, pre[16];
#pragma unroll
      for (int t = 0; t < 16; ++t) {
        e[t] = (j0 + t <= ig) ? __expf(x[t] - mx) : 0.f;
        c += e[t];
        pre[t] = c;  // inclusive local prefix
      }
      float tot = c;
      for (int off = 16; off; off >>= 1) tot += __shfl_xor(tot, off, 32);
      const float inv = 1.f / tot;
      float scn = c;  // exclusive cross-lane scan of lane totals
      for (int off = 1; off < 32; off <<= 1) {
        const float v = __shfl_up(scn, off, 32);
        if (lane >= off) scn += v;
      }
      const float excl = scn - c;

      float ns[16];
#pragma unroll
      for (int t = 0; t < 16; ++t) {
        const float cum = (excl + pre[t]) * inv;
        const float jd  = fabsf((float)(ig - (j0 + t)));
        const float dist = sqrtf(fmaxf((1.f - cum) * jd, 0.f));
        float eff = __expf(dist * gamma);
        eff = fminf(fmaxf(eff, 1e-5f), 1e5f);
        ns[t] = (j0 + t <= ig) ? x[t] * eff : -3.0e38f;
      }
      float mx2 = -3.0e38f;
#pragma unroll
      for (int t = 0; t < 16; ++t) mx2 = fmaxf(mx2, ns[t]);
      for (int off = 16; off; off >>= 1) mx2 = fmaxf(mx2, __shfl_xor(mx2, off, 32));
      float p[16], s2 = 0.f;
#pragma unroll
      for (int t = 0; t < 16; ++t) {
        p[t] = (j0 + t <= ig) ? __expf(ns[t] - mx2) : 0.f;
        s2 += p[t];
      }
      for (int off = 16; off; off >>= 1) s2 += __shfl_xor(s2, off, 32);
      const float inv2 = (ig == 0) ? 0.f : (1.f / s2);  // row 0 zero-padded
#pragma unroll
      for (int t = 0; t < 16; ++t) row[j0 + t] = p[t] * inv2;
    }
  }

  // ---- phase 3: out_mean = P @ vm, out_cov = P^2 @ vc ----
  v8f om0 = {0.f, 0.f, 0.f, 0.f, 0.f, 0.f, 0.f, 0.f};
  v8f om1 = om0, oc0 = om0, oc1 = om0;
  const int vkey = tid >> 3, vd0 = (tid & 7) * 8;  // transposed staging coords
  for (int kt = 0; kt < 16; ++kt) {
    __syncthreads();
    {
      const __bf16* pm = VM + headOff + (size_t)(kt * 32 + vkey) * 64 + vd0;
      const __bf16* pc = VC + headOff + (size_t)(kt * 32 + vkey) * 64 + vd0;
#pragma unroll
      for (int j = 0; j < 8; ++j) {  // write transposed: [d][key]
        lK1[(vd0 + j) * VLD + vkey] = pm[j];
        lK2[(vd0 + j) * VLD + vkey] = pc[j];
      }
    }
    __syncthreads();
    v16bf aP  = ld_a_f32(S, mrow, SLD, kt * 32, lane, false);
    v16bf aP2 = ld_a_f32(S, mrow, SLD, kt * 32, lane, true);
    v16bf bM0 = ld_b_bf16(lK1, ncol,      VLD, 0, lane);
    v16bf bM1 = ld_b_bf16(lK1, ncol + 16, VLD, 0, lane);
    v16bf bC0 = ld_b_bf16(lK2, ncol,      VLD, 0, lane);
    v16bf bC1 = ld_b_bf16(lK2, ncol + 16, VLD, 0, lane);
    om0 = WMMA_BF16(aP, bM0, om0);
    om1 = WMMA_BF16(aP, bM1, om1);
    oc0 = WMMA_BF16(aP2, bC0, oc0);
    oc1 = WMMA_BF16(aP2, bC1, oc1);
  }
  // write concat layout [b, s, h*64+d] as bf16
#pragma unroll
  for (int nt = 0; nt < 2; ++nt) {
    const v8f& am = nt ? om1 : om0;
    const v8f& ac = nt ? oc1 : oc0;
#pragma unroll
    for (int r = 0; r < 8; ++r) {
      const int s = rb * 64 + mrow + r + hi8;
      const int d = ncol + nt * 16 + ln;
      const size_t o = ((size_t)b * 512 + s) * 512 + h * 64 + d;
      CM[o] = (__bf16)am[r];
      CC[o] = (__bf16)ac[r];
    }
  }
}

// ---------------------------------------------------------------------------
// Host-side orchestration
// ---------------------------------------------------------------------------
extern "C" void kernel_launch(void* const* d_in, const int* in_sizes, int n_in,
                              void* d_out, int out_size, void* d_ws, size_t ws_size,
                              hipStream_t stream) {
  (void)in_sizes; (void)n_in; (void)out_size; (void)ws_size;
  char* ws = (char*)d_ws;
  const size_t PROJ = (size_t)16 * 8 * 512 * 64;  // elements per projected tensor
  __bf16* qm = (__bf16*)ws;
  __bf16* qc = qm + PROJ;
  __bf16* km = qc + PROJ;
  __bf16* kc = km + PROJ;
  __bf16* vm = kc + PROJ;
  __bf16* vc = vm + PROJ;
  float* rs_qm = (float*)(ws + 6 * PROJ * sizeof(__bf16));
  float* rs_qc = rs_qm + 65536;
  float* rs_km = rs_qc + 65536;
  float* rs_kc = rs_km + 65536;
  __bf16* cm = (__bf16*)(rs_kc + 65536);
  __bf16* cc = cm + (size_t)16 * 512 * 512;

  const dim3 tb(256);
  const dim3 gG(128, 8);  // 8192/64 x 512/64

  // 1) input projections -> bf16 [b,h,s,d]
  gemm_awt_kernel<false, true><<<gG, tb, 0, stream>>>(d_in[0], (const float*)d_in[6],  qm);
  gemm_awt_kernel<false, true><<<gG, tb, 0, stream>>>(d_in[1], (const float*)d_in[7],  qc);
  gemm_awt_kernel<false, true><<<gG, tb, 0, stream>>>(d_in[2], (const float*)d_in[8],  km);
  gemm_awt_kernel<false, true><<<gG, tb, 0, stream>>>(d_in[3], (const float*)d_in[9],  kc);
  gemm_awt_kernel<false, true><<<gG, tb, 0, stream>>>(d_in[4], (const float*)d_in[10], vm);
  gemm_awt_kernel<false, true><<<gG, tb, 0, stream>>>(d_in[5], (const float*)d_in[11], vc);

  // 2) row sums (+ in-place sqrt-clip for cov operands)
  stats_kernel<<<8192, tb, 0, stream>>>(qm, rs_qm, 0);
  stats_kernel<<<8192, tb, 0, stream>>>(qc, rs_qc, 1);
  stats_kernel<<<8192, tb, 0, stream>>>(km, rs_km, 0);
  stats_kernel<<<8192, tb, 0, stream>>>(kc, rs_kc, 1);

  // 3) fused attention: 16 b x 8 h x 8 row-panels, 168960 B dynamic LDS
  attn_kernel<<<dim3(1024), tb, 168960, stream>>>(qm, qc, km, kc, vm, vc,
                                                  rs_qm, rs_qc, rs_km, rs_kc,
                                                  (const float*)d_in[14], cm, cc);

  // 4) output projections -> f32 d_out halves
  gemm_awt_kernel<true, false><<<gG, tb, 0, stream>>>(cm, (const float*)d_in[12],
                                                      (float*)d_out);
  gemm_awt_kernel<true, false><<<gG, tb, 0, stream>>>(cc, (const float*)d_in[13],
                                                      (float*)d_out + (size_t)8192 * 512);
}
